// RecurrentGCN_27376121544893
// MI455X (gfx1250) — compile-verified, matrix operand order checked by reading, not run on hardware
//
#include <hip/hip_runtime.h>
#include <hip/hip_bf16.h>

typedef __bf16 bf16_t;
typedef bf16_t v16bf __attribute__((ext_vector_type(16)));
typedef float  v8f   __attribute__((ext_vector_type(8)));

#if __has_builtin(__builtin_amdgcn_global_load_async_to_lds_b128)
#define HAVE_ASYNC_LDS 1
typedef int v4i_async __attribute__((ext_vector_type(4)));
#else
#define HAVE_ASYNC_LDS 0
#endif

__device__ __forceinline__ float sigmoidf_(float x) { return 1.0f / (1.0f + __expf(-x)); }

__device__ __forceinline__ void atomAddF(float* p, float v) {
  __hip_atomic_fetch_add(p, v, __ATOMIC_RELAXED, __HIP_MEMORY_SCOPE_AGENT);
}

// deg[src] += w  (segment_sum by src)
__global__ void k_deg(const int* __restrict__ ei, const float* __restrict__ w,
                      float* __restrict__ deg, int E) {
  int e = blockIdx.x * blockDim.x + threadIdx.x;
  if (e >= E) return;
  atomAddF(&deg[ei[e]], w[e]);
}

// deg -> d^{-1/2} in place
__global__ void k_dinv(float* __restrict__ deg, int N) {
  int i = blockIdx.x * blockDim.x + threadIdx.x;
  if (i >= N) return;
  float d = deg[i];
  deg[i] = (d > 0.0f) ? rsqrtf(fmaxf(d, 1e-12f)) : 0.0f;
}

// norm[e] = -dinv[src] * w * dinv[dst]
__global__ void k_norm(const int* __restrict__ ei, const float* __restrict__ w,
                       const float* __restrict__ dinv, float* __restrict__ normv, int E) {
  int e = blockIdx.x * blockDim.x + threadIdx.x;
  if (e >= E) return;
  int s = ei[e], d = ei[E + e];
  normv[e] = -dinv[s] * w[e] * dinv[d];
}

// Y[dst] += norm * X[src], 32 features -> one wave per edge, lane = feature
__global__ void k_prop(const float* __restrict__ X, const int* __restrict__ ei,
                       const float* __restrict__ normv, float* __restrict__ Y, int E) {
  long tid = (long)blockIdx.x * blockDim.x + threadIdx.x;
  long e = tid >> 5;
  int  f = (int)(tid & 31);
  if (e >= E) return;
  if (f == 0 && (e + 2048) < (long)E) {
    __builtin_prefetch(&normv[e + 2048], 0, 0);
    __builtin_prefetch(&ei[e + 2048], 0, 0);
  }
  int s = ei[e];
  int d = ei[(long)E + e];
  float v = normv[e] * X[(long)s * 32 + f];
  atomAddF(&Y[(long)d * 32 + f], v);
}

// Tcat[:, col*32 + j] = (bf16) T[:, j]
__global__ void k_cast(const float* __restrict__ T, bf16_t* __restrict__ Tcat, int col, int N) {
  long tid = (long)blockIdx.x * blockDim.x + threadIdx.x;
  if (tid >= (long)N * 32) return;
  long m = tid >> 5; int j = (int)(tid & 31);
  Tcat[m * 160 + col * 32 + j] = (bf16_t)T[tid];
}

// P = 2*P - Tprev (in place), also write bf16 into Tcat column block
__global__ void k_cheb(float* __restrict__ P, const float* __restrict__ Tprev,
                       bf16_t* __restrict__ Tcat, int col, int N) {
  long tid = (long)blockIdx.x * blockDim.x + threadIdx.x;
  if (tid >= (long)N * 32) return;
  long m = tid >> 5; int j = (int)(tid & 31);
  float v = 2.0f * P[tid] - Tprev[tid];
  P[tid] = v;
  Tcat[m * 160 + col * 32 + j] = (bf16_t)v;
}

// Pack weights into WMMA B-fragment layout + fold biases.
// Fragment (nt,kt): 32 lanes x 16 bf16 contiguous. Lane l holds column
// n = nt*16 + (l&15); lanes<16 carry k_local {0..7,16..23}, lanes>=16 {8..15,24..31}.
__global__ void k_pack(const float* __restrict__ Wx, const float* __restrict__ bx,
                       const float* __restrict__ bh, const float* __restrict__ bb,
                       bf16_t* __restrict__ Bfrag, float* __restrict__ biasvec) {
  const int gmap[3] = {0, 2, 3};  // gates i, c, o (forget gate is dead: F*C==0)
  int t = blockIdx.x * blockDim.x + threadIdx.x;
  if (t < 6 * 5 * 32 * 16) {
    int s = t & 15, l = (t >> 4) & 31, frag = t >> 9;
    int kt = frag % 5, nt = frag / 5;
    int kl = (l < 16) ? ((s < 8) ? s : s + 8) : ((s < 8) ? s + 8 : s + 16);
    int n = nt * 16 + (l & 15);
    int g = gmap[n >> 5], j = n & 31;
    // Wx layout [4,5,32,32]: ((g*5 + kcheb)*32 + in)*32 + out
    Bfrag[t] = (bf16_t)Wx[((g * 5 + kt) * 32 + kl) * 32 + j];
  } else if (t < 6 * 5 * 32 * 16 + 96) {
    int c = t - 6 * 5 * 32 * 16;
    int g = gmap[c >> 5], j = c & 31;
    biasvec[c] = bx[g * 32 + j] + bh[g * 32 + j] + bb[g * 32 + j];
  }
}

// G[N,96] = Tcat[N,160](bf16) @ Wcat[160,96](bf16 fragments), f32 accumulate.
// Block: 192 threads = 6 waves, wave w owns N-tile w; grid.x = M tiles.
__global__ void __launch_bounds__(192) k_gemm(const bf16_t* __restrict__ Tcat,
                                              const bf16_t* __restrict__ Bfrag,
                                              float* __restrict__ G, int N) {
  __shared__ __align__(16) bf16_t As[16 * 160];
  long m0 = (long)blockIdx.x * 16;
  const uint4* srcq = (const uint4*)(Tcat + m0 * 160);
  uint4* dstq = (uint4*)As;
  bool full = (m0 + 16 <= (long)N);

#if HAVE_ASYNC_LDS
  if (full) {
    // Async copy 16 rows x 320 B = 320 x b128 chunks, ASYNCcnt-tracked.
    for (int i = threadIdx.x; i < 320; i += 192) {
      __builtin_amdgcn_global_load_async_to_lds_b128(
          (v4i_async*)(uintptr_t)(srcq + i),
          (v4i_async*)(dstq + i),
          0, 0);
    }
    asm volatile("s_wait_asynccnt 0x0" ::: "memory");
    __syncthreads();
  } else
#endif
  {
    for (int i = threadIdx.x; i < 320; i += 192) {
      int row = i / 20;
      uint4 z = {0u, 0u, 0u, 0u};
      dstq[i] = ((m0 + row) < N) ? srcq[i] : z;
    }
    __syncthreads();
  }

  int wave = threadIdx.x >> 5;   // ntile 0..5
  int lane = threadIdx.x & 31;
  int hi   = lane >> 4;
  int lrow = lane & 15;
  v8f acc = {0.f, 0.f, 0.f, 0.f, 0.f, 0.f, 0.f, 0.f};
#pragma unroll
  for (int kt = 0; kt < 5; ++kt) {
    union { v16bf v; uint4 q[2]; } A, B;
    // A per ISA layout: lanes<16 k {0..7,16..23}, lanes>=16 k {8..15,24..31}
    const uint4* aq = (const uint4*)(As + lrow * 160 + kt * 32 + hi * 8);
    A.q[0] = aq[0];
    A.q[1] = aq[2];
    const uint4* bq = (const uint4*)(Bfrag + (((wave * 5 + kt) * 32 + lane) << 4));
    B.q[0] = bq[0];
    B.q[1] = bq[1];
    acc = __builtin_amdgcn_wmma_f32_16x16x32_bf16(false, A.v, false, B.v,
                                                  (short)0, acc, false, false);
  }
  int n = wave * 16 + lrow;
  if (full) {
    float* gp = G + (m0 + hi * 8) * 96 + n;
#pragma unroll
    for (int r = 0; r < 8; ++r) gp[(long)r * 96] = acc[r];
  } else {
#pragma unroll
    for (int r = 0; r < 8; ++r) {
      long m = m0 + r + hi * 8;
      if (m < N) G[m * 96 + n] = acc[r];
    }
  }
}

// Gates -> h = relu(O * tanh(Cn)); I=σ(gi), Cn=I*tanh(gc), O=σ(go + wc2*Cn)
__global__ void k_lstm(const float* __restrict__ G, const float* __restrict__ bias,
                       const float* __restrict__ wc, float* __restrict__ H, int N) {
  long tid = (long)blockIdx.x * blockDim.x + threadIdx.x;
  if (tid >= (long)N * 32) return;
  long m = tid >> 5; int j = (int)(tid & 31);
  const float* g = G + m * 96;
  float gi = g[j]      + bias[j];
  float gc = g[32 + j] + bias[32 + j];
  float go = g[64 + j] + bias[64 + j];
  float I  = sigmoidf_(gi);
  float Cn = I * tanhf(gc);
  float O  = sigmoidf_(go + wc[64 + j] * Cn);   // wc row 2
  H[tid] = fmaxf(0.0f, O * tanhf(Cn));
}

// Collapse 4 bias-only Linears: wEff[0..31] = hw1@hw2@hw3@hw4, wEff[32] = folded bias
__global__ void k_headprep(const float* __restrict__ hw1, const float* __restrict__ hb1,
                           const float* __restrict__ hw2, const float* __restrict__ hb2,
                           const float* __restrict__ hw3, const float* __restrict__ hb3,
                           const float* __restrict__ hw4, const float* __restrict__ hb4,
                           float* __restrict__ wEff) {
  int i = threadIdx.x;
  if (i < 32) {
    float acc = 0.0f;
    for (int a = 0; a < 16; ++a)
      for (int b = 0; b < 8; ++b)
        for (int c = 0; c < 4; ++c)
          acc += hw1[i * 16 + a] * hw2[a * 8 + b] * hw3[b * 4 + c] * hw4[c];
    wEff[i] = acc;
  } else if (i == 32) {
    float b2[8], b3[4];
    for (int b = 0; b < 8; ++b) {
      float s = hb2[b];
      for (int a = 0; a < 16; ++a) s += hb1[a] * hw2[a * 8 + b];
      b2[b] = s;
    }
    for (int c = 0; c < 4; ++c) {
      float s = hb3[c];
      for (int b = 0; b < 8; ++b) s += b2[b] * hw3[b * 4 + c];
      b3[c] = s;
    }
    float s = hb4[0];
    for (int c = 0; c < 4; ++c) s += b3[c] * hw4[c];
    wEff[32] = s;
  }
}

__global__ void k_head(const float* __restrict__ H, const float* __restrict__ wEff,
                       float* __restrict__ out, int N) {
  int m = blockIdx.x * blockDim.x + threadIdx.x;
  if (m >= N) return;
  float s = wEff[32];
  const float* h = H + (long)m * 32;
#pragma unroll
  for (int j = 0; j < 32; ++j) s += h[j] * wEff[j];
  out[m] = s;
}

extern "C" void kernel_launch(void* const* d_in, const int* in_sizes, int n_in,
                              void* d_out, int out_size, void* d_ws, size_t ws_size,
                              hipStream_t stream) {
  (void)n_in; (void)out_size; (void)ws_size;
  const float* x  = (const float*)d_in[0];
  const int*   ei = (const int*)d_in[1];
  const float* w  = (const float*)d_in[2];
  const float* Wx[2] = {(const float*)d_in[3], (const float*)d_in[9]};
  const float* bx[2] = {(const float*)d_in[4], (const float*)d_in[10]};
  const float* bh[2] = {(const float*)d_in[6], (const float*)d_in[12]};
  const float* wc[2] = {(const float*)d_in[7], (const float*)d_in[13]};
  const float* bb[2] = {(const float*)d_in[8], (const float*)d_in[14]};
  const float* hw1 = (const float*)d_in[15]; const float* hb1 = (const float*)d_in[16];
  const float* hw2 = (const float*)d_in[17]; const float* hb2 = (const float*)d_in[18];
  const float* hw3 = (const float*)d_in[19]; const float* hb3 = (const float*)d_in[20];
  const float* hw4 = (const float*)d_in[21]; const float* hb4 = (const float*)d_in[22];

  int N = in_sizes[0] / 32;
  int E = in_sizes[2];

  char* ws = (char*)d_ws;
  size_t off = 0;
  auto take = [&](size_t bytes) -> char* {
    char* p = ws + off;
    off = (off + bytes + 255) & ~(size_t)255;
    return p;
  };
  float*  normv   = (float*)take((size_t)E * 4);
  float*  deg     = (float*)take((size_t)N * 4);
  float*  B0      = (float*)take((size_t)N * 128);
  float*  B1      = (float*)take((size_t)N * 128);
  float*  B2      = (float*)take((size_t)N * 128);
  float*  Hbuf    = (float*)take((size_t)N * 128);
  bf16_t* Tcat    = (bf16_t*)take((size_t)N * 320);
  bf16_t* Bfrag   = (bf16_t*)take((size_t)(6 * 5 * 512) * 2);
  float*  biasvec = (float*)take(96 * 4);
  float*  G       = (float*)take((size_t)N * 96 * 4);
  float*  wEff    = (float*)take(33 * 4);

  int gE   = (E + 255) / 256;
  int gN   = (N + 255) / 256;
  int gN32 = (int)(((long)N * 32 + 255) / 256);
  int gE32 = (int)(((long)E * 32 + 255) / 256);

  // Graph normalization (once; both layers share it)
  (void)hipMemsetAsync(deg, 0, (size_t)N * 4, stream);
  k_deg <<<gE, 256, 0, stream>>>(ei, w, deg, E);
  k_dinv<<<gN, 256, 0, stream>>>(deg, N);
  k_norm<<<gE, 256, 0, stream>>>(ei, w, deg, normv, E);

  const float* Xin = x;
  float* bufs[3] = {B0, B1, B2};
  for (int L = 0; L < 2; ++L) {
    k_pack<<<(6 * 5 * 32 * 16 + 96 + 255) / 256, 256, 0, stream>>>(
        Wx[L], bx[L], bh[L], bb[L], Bfrag, biasvec);
    // T0 = X
    k_cast<<<gN32, 256, 0, stream>>>(Xin, Tcat, 0, N);
    // T1 = L_hat X
    (void)hipMemsetAsync(B0, 0, (size_t)N * 128, stream);
    k_prop<<<gE32, 256, 0, stream>>>(Xin, ei, normv, B0, E);
    k_cast<<<gN32, 256, 0, stream>>>(B0, Tcat, 1, N);
    // Tk = 2 L Tk-1 - Tk-2
    const float* Tp = Xin;
    float* Tc = B0;
    for (int k = 2; k <= 4; ++k) {
      float* P = bufs[(k - 1) % 3];
      (void)hipMemsetAsync(P, 0, (size_t)N * 128, stream);
      k_prop<<<gE32, 256, 0, stream>>>(Tc, ei, normv, P, E);
      k_cheb<<<gN32, 256, 0, stream>>>(P, Tp, Tcat, k, N);
      Tp = Tc; Tc = P;
    }
    // All three live gates in one WMMA GEMM, then pointwise LSTM + relu
    k_gemm<<<(N + 15) / 16, 192, 0, stream>>>(Tcat, Bfrag, G, N);
    k_lstm<<<gN32, 256, 0, stream>>>(G, biasvec, wc[L], Hbuf, N);
    Xin = Hbuf;
  }
  k_headprep<<<1, 64, 0, stream>>>(hw1, hb1, hw2, hb2, hw3, hb3, hw4, hb4, wEff);
  k_head<<<gN, 256, 0, stream>>>(Hbuf, wEff, (float*)d_out, N);
}
